// ModelEmbedding_GAT2_90263032693071
// MI455X (gfx1250) — matrix operand order repaired and checked by reading, not compiled, not run on hardware
//
#include <hip/hip_runtime.h>
#include <hip/hip_bf16.h>
#include <math.h>

#define NN   20000
#define EE   180000
#define BB   16
#define SS   512
#define DD   768
#define F1   128      // HEADS*HID
#define NEG_SLOPE 0.2f

typedef _Float16 v16h __attribute__((ext_vector_type(16)));
typedef float    v8f  __attribute__((ext_vector_type(8)));

// ---- monotone uint encoding of float for atomic max ----
__device__ __forceinline__ unsigned fenc(float f) {
  unsigned u = __float_as_uint(f);
  return (u & 0x80000000u) ? ~u : (u | 0x80000000u);
}
__device__ __forceinline__ float fdec(unsigned e) {
  return __uint_as_float((e & 0x80000000u) ? (e ^ 0x80000000u) : ~e);
}
#define ENC_NEG_INF 0x007FFFFFu   // fenc(-inf)

__global__ void fill_u32(unsigned* __restrict__ p, unsigned v, int n) {
  int i = blockIdx.x * blockDim.x + threadIdx.x;
  if (i < n) p[i] = v;
}

// ---------- Layer 1: projections (N,4) x (4,128) -> hs1, hd1 ----------
__global__ void l1_proj(const float* __restrict__ feat,
                        const float* __restrict__ Ws, const float* __restrict__ Wd,
                        float* __restrict__ hs, float* __restrict__ hd) {
  int n = blockIdx.x;          // node
  int o = threadIdx.x;         // 0..127
  float f0 = feat[n * 4 + 0], f1 = feat[n * 4 + 1];
  float f2 = feat[n * 4 + 2], f3 = feat[n * 4 + 3];
  float s = f0 * Ws[o] + f1 * Ws[128 + o] + f2 * Ws[256 + o] + f3 * Ws[384 + o];
  float d = f0 * Wd[o] + f1 * Wd[128 + o] + f2 * Wd[256 + o] + f3 * Wd[384 + o];
  hs[(long)n * F1 + o] = s;
  hd[(long)n * F1 + o] = d;
}

// ---------- Layer 1: edge scores + segment max ----------
__global__ void l1_score(const int* __restrict__ es, const int* __restrict__ ed,
                         const float* __restrict__ hs, const float* __restrict__ hd,
                         const float* __restrict__ attn,
                         float* __restrict__ score, unsigned* __restrict__ menc) {
  long idx = (long)blockIdx.x * blockDim.x + threadIdx.x;
  if (idx >= (long)EE * 8) return;
  int e = (int)(idx >> 3), h = (int)(idx & 7);
  int s = es[e], d = ed[e];
  const float* ps = hs + (long)s * F1 + h * 16;
  const float* pd = hd + (long)d * F1 + h * 16;
  float acc = 0.f;
#pragma unroll
  for (int f = 0; f < 16; f++) {
    float t = ps[f] + pd[f];
    t = t > 0.f ? t : NEG_SLOPE * t;
    acc += attn[h * 16 + f] * t;
  }
  score[(long)e * 8 + h] = acc;
  atomicMax(&menc[d * 8 + h], fenc(acc));
}

// ---------- Layer 1: exp + segment sum ----------
__global__ void l1_exp(const int* __restrict__ ed, float* __restrict__ score_ex,
                       const unsigned* __restrict__ menc, float* __restrict__ denom) {
  long idx = (long)blockIdx.x * blockDim.x + threadIdx.x;
  if (idx >= (long)EE * 8) return;
  int e = (int)(idx >> 3), h = (int)(idx & 7);
  int d = ed[e];
  float ex = expf(score_ex[(long)e * 8 + h] - fdec(menc[d * 8 + h]));
  score_ex[(long)e * 8 + h] = ex;
  atomicAdd(&denom[d * 8 + h], ex);
}

// ---------- Layer 1: alpha-weighted aggregation ----------
__global__ void l1_agg(const int* __restrict__ es, const int* __restrict__ ed,
                       const float* __restrict__ hs, const float* __restrict__ ex,
                       const float* __restrict__ denom, float* __restrict__ h1) {
  long idx = (long)blockIdx.x * blockDim.x + threadIdx.x;
  if (idx >= (long)EE * F1) return;
  int e = (int)(idx >> 7), o = (int)(idx & 127), h = o >> 4;
  int s = es[e], d = ed[e];
  float alpha = ex[(long)e * 8 + h] / denom[d * 8 + h];
  atomicAdd(&h1[(long)d * F1 + o], alpha * hs[(long)s * F1 + o]);
}

// ---------- Layer 1: ELU + pack into WMMA A-fragment order (f16) ----------
// A 16-bit 16x32 layout (ISA 7.12.2): lane = sel*16 + m; within-lane half index
// j covers K = kblk*32 + sel*8 + {0..7, 16..23}.
__global__ void l1_elu_packA(const float* __restrict__ h1, _Float16* __restrict__ Apk) {
  long idx = (long)blockIdx.x * blockDim.x + threadIdx.x;
  if (idx >= (long)NN * F1) return;
  int n = (int)(idx / F1), k = (int)(idx & (F1 - 1));
  float x = h1[idx];
  float y = x > 0.f ? x : expm1f(x);
  int rt = n >> 4, ml = n & 15;
  int kb = k >> 5, kin = k & 31;
  int sel = (kin >> 3) & 1;
  int j = (kin & 7) | ((kin >> 4) << 3);
  long dst = (((long)rt * 4 + kb) * 32 + sel * 16 + ml) * (long)16 + j;
  Apk[dst] = (_Float16)y;
}

// ---------- Weights: f32 -> f16 packed B-fragment order ----------
// B layout: lane = sel*16 + n (column); half j covers K = kblk*32 + sel*8 + {0..7, 16..23}.
__global__ void pack_B(const float* __restrict__ w, _Float16* __restrict__ wp) {
  int i = blockIdx.x * blockDim.x + threadIdx.x;
  if (i >= F1 * DD) return;
  int k = i / DD, ng = i % DD;
  int t = ng >> 4, n = ng & 15;
  int kb = k >> 5, kin = k & 31;
  int sel = (kin >> 3) & 1, j = (kin & 7) | ((kin >> 4) << 3);
  long dst = ((((long)t * 4 + kb) * 2 + sel) * 16 + n) * (long)16 + j;
  wp[dst] = (_Float16)w[i];
}

// ---------- Layer 2 projections: (N,128)f16 @ (128,768)f16 -> f32, twice ----------
// One wave computes a 16x64 strip (4 col tiles), A fragment held in registers
// across 8 WMMAs per K-block. All fragment loads are contiguous 32B per lane.
__global__ void l2_gemm(const _Float16* __restrict__ Apk,
                        const _Float16* __restrict__ Bs, const _Float16* __restrict__ Bd,
                        float* __restrict__ Cs, float* __restrict__ Cd) {
  int rowtile = blockIdx.x;                                  // 0..1249
  int strip   = blockIdx.y * blockDim.y + threadIdx.y;       // 0..11
  int lane    = threadIdx.x;                                 // 0..31
  int ml  = lane & 15;
  int sel = lane >> 4;
  v8f accS[4] = {};
  v8f accD[4] = {};
#pragma unroll
  for (int kb = 0; kb < 4; kb++) {
    const v16h a = *(const v16h*)(Apk + (((long)rowtile * 4 + kb) * 32 + lane) * 16);
#pragma unroll
    for (int c = 0; c < 4; c++) {
      int tile = strip * 4 + c;
      long bbase = ((((long)tile * 4 + kb) * 2 + sel) * 16 + ml) * (long)16;
      v16h bs = *(const v16h*)(Bs + bbase);
      v16h bd = *(const v16h*)(Bd + bbase);
      accS[c] = __builtin_amdgcn_wmma_f32_16x16x32_f16(false, a, false, bs, (short)0, accS[c], false, false);
      accD[c] = __builtin_amdgcn_wmma_f32_16x16x32_f16(false, a, false, bd, (short)0, accD[c], false, false);
    }
  }
#pragma unroll
  for (int c = 0; c < 4; c++) {
#pragma unroll
    for (int v = 0; v < 8; v++) {
      int r = rowtile * 16 + sel * 8 + v;    // C/D layout: VGPR v -> M = v (+8 upper lanes)
      int col = (strip * 4 + c) * 16 + ml;
      Cs[(long)r * DD + col] = accS[c][v];
      Cd[(long)r * DD + col] = accD[c][v];
    }
  }
}

// ---------- Layer 2: edge scores (wave per edge) ----------
__global__ void l2_score(const int* __restrict__ es, const int* __restrict__ ed,
                         const float* __restrict__ hs2, const float* __restrict__ hd2,
                         const float* __restrict__ attn2,
                         float* __restrict__ score, unsigned* __restrict__ menc) {
  int e = blockIdx.x * blockDim.y + threadIdx.y;
  if (e >= EE) return;
  int s = es[e], d = ed[e];
  const float* ps = hs2 + (long)s * DD;
  const float* pd = hd2 + (long)d * DD;
  float acc = 0.f;
  for (int i = threadIdx.x; i < DD; i += 32) {
    float t = ps[i] + pd[i];
    t = t > 0.f ? t : NEG_SLOPE * t;
    acc += attn2[i] * t;
  }
#pragma unroll
  for (int off = 16; off > 0; off >>= 1) acc += __shfl_xor(acc, off, 32);
  if (threadIdx.x == 0) {
    score[e] = acc;
    atomicMax(&menc[d], fenc(acc));
  }
}

__global__ void l2_exp(const int* __restrict__ ed, float* __restrict__ score_ex,
                       const unsigned* __restrict__ menc, float* __restrict__ denom) {
  int e = blockIdx.x * blockDim.x + threadIdx.x;
  if (e >= EE) return;
  int d = ed[e];
  float ex = expf(score_ex[e] - fdec(menc[d]));
  score_ex[e] = ex;
  atomicAdd(&denom[d], ex);
}

__global__ void l2_agg(const int* __restrict__ es, const int* __restrict__ ed,
                       const float* __restrict__ hs2, const float* __restrict__ ex,
                       const float* __restrict__ denom, float* __restrict__ nodeE) {
  int e = blockIdx.x * blockDim.y + threadIdx.y;
  if (e >= EE) return;
  int s = es[e], d = ed[e];
  float alpha = ex[e] / denom[d];
  const float* ps = hs2 + (long)s * DD;
  float* po = nodeE + (long)d * DD;
  for (int i = threadIdx.x; i < DD; i += 32)
    atomicAdd(&po[i], alpha * ps[i]);
}

// ---------- Final assembly: pe + token + time + grid ----------
__global__ void assemble(const int* __restrict__ tok, const int* __restrict__ minl,
                         const int* __restrict__ wdl, const int* __restrict__ dayl,
                         const int* __restrict__ gridl,
                         const float* __restrict__ nodeE,
                         const float* __restrict__ gridT, const float* __restrict__ dtT,
                         const float* __restrict__ wkT, const float* __restrict__ dayT,
                         float* __restrict__ out) {
  long idx = (long)blockIdx.x * blockDim.x + threadIdx.x;
  if (idx >= (long)BB * SS * DD) return;
  int dch  = (int)(idx % DD);
  long bs  = idx / DD;                 // b*512+s
  int  s   = (int)(bs % SS);
  int flat = (int)bs;
  int t = tok[flat];
  float val = nodeE[(long)t * DD + dch];
  val += wkT[(long)wdl[flat] * DD + dch];
  val += dtT[(long)minl[flat] * DD + dch];
  val += dayT[(long)dayl[flat] * DD + dch];
  val += gridT[(long)gridl[flat] * DD + dch];
  // positional embedding, computed on the fly
  int i2 = dch & ~1;
  float div = __expf(-(float)i2 * (9.210340371976184f / (float)DD));
  float ang = (float)s * div;
  val += (dch & 1) ? __cosf(ang) : __sinf(ang);
  out[idx] = val;
}

extern "C" void kernel_launch(void* const* d_in, const int* in_sizes, int n_in,
                              void* d_out, int out_size, void* d_ws, size_t ws_size,
                              hipStream_t stream) {
  const int*   trj_token = (const int*)d_in[0];
  const int*   min_list  = (const int*)d_in[1];
  const int*   wd_list   = (const int*)d_in[2];
  const int*   day_list  = (const int*)d_in[3];
  const int*   grid_list = (const int*)d_in[4];
  /* d_in[5] = poi_list: unused by the reference */
  const int*   edge_src  = (const int*)d_in[6];
  const int*   edge_dst  = (const int*)d_in[7];
  const float* feature   = (const float*)d_in[8];
  const float* W_src1    = (const float*)d_in[9];
  const float* W_dst1    = (const float*)d_in[10];
  const float* attn1     = (const float*)d_in[11];
  const float* W_src2    = (const float*)d_in[12];
  const float* W_dst2    = (const float*)d_in[13];
  const float* attn2     = (const float*)d_in[14];
  const float* grid_tbl  = (const float*)d_in[15];
  const float* dt_tbl    = (const float*)d_in[16];
  const float* wk_tbl    = (const float*)d_in[17];
  const float* day_tbl   = (const float*)d_in[18];
  float* out = (float*)d_out;

  // ---- carve workspace ----
  char* p = (char*)d_ws;
  auto carve = [&](size_t bytes) -> void* {
    void* r = (void*)p;
    p += (bytes + 255) & ~(size_t)255;
    return r;
  };
  float*     hs1    = (float*)carve((size_t)NN * F1 * 4);
  float*     hd1    = (float*)carve((size_t)NN * F1 * 4);
  float*     sc1    = (float*)carve((size_t)EE * 8  * 4);   // score -> ex (in place)
  unsigned*  m1     = (unsigned*)carve((size_t)NN * 8 * 4);
  float*     den1   = (float*)carve((size_t)NN * 8 * 4);
  float*     h1acc  = (float*)carve((size_t)NN * F1 * 4);
  _Float16*  Apk    = (_Float16*)carve((size_t)NN * F1 * 2);
  _Float16*  Ws2p   = (_Float16*)carve((size_t)F1 * DD * 2);
  _Float16*  Wd2p   = (_Float16*)carve((size_t)F1 * DD * 2);
  float*     hs2    = (float*)carve((size_t)NN * DD * 4);
  float*     hd2    = (float*)carve((size_t)NN * DD * 4);
  float*     sc2    = (float*)carve((size_t)EE * 4);        // score -> ex (in place)
  unsigned*  m2     = (unsigned*)carve((size_t)NN * 4);
  float*     den2   = (float*)carve((size_t)NN * 4);
  float*     nodeE  = (float*)carve((size_t)NN * DD * 4);

  const int T = 256;
  // ---- init accumulators (must be re-done every call: graph replays) ----
  fill_u32<<<(NN * 8 + T - 1) / T, T, 0, stream>>>(m1, ENC_NEG_INF, NN * 8);
  fill_u32<<<(NN * 8 + T - 1) / T, T, 0, stream>>>((unsigned*)den1, 0u, NN * 8);
  fill_u32<<<(NN * F1 + T - 1) / T, T, 0, stream>>>((unsigned*)h1acc, 0u, NN * F1);
  fill_u32<<<(NN + T - 1) / T, T, 0, stream>>>(m2, ENC_NEG_INF, NN);
  fill_u32<<<(NN + T - 1) / T, T, 0, stream>>>((unsigned*)den2, 0u, NN);
  fill_u32<<<(NN * DD + T - 1) / T, T, 0, stream>>>((unsigned*)nodeE, 0u, NN * DD);

  // ---- layer 1 ----
  l1_proj<<<NN, F1, 0, stream>>>(feature, W_src1, W_dst1, hs1, hd1);
  {
    long n = (long)EE * 8;
    l1_score<<<(unsigned)((n + T - 1) / T), T, 0, stream>>>(edge_src, edge_dst, hs1, hd1, attn1, sc1, m1);
    l1_exp<<<(unsigned)((n + T - 1) / T), T, 0, stream>>>(edge_dst, sc1, m1, den1);
  }
  {
    long n = (long)EE * F1;
    l1_agg<<<(unsigned)((n + T - 1) / T), T, 0, stream>>>(edge_src, edge_dst, hs1, sc1, den1, h1acc);
  }
  l1_elu_packA<<<(NN * F1 + T - 1) / T, T, 0, stream>>>(h1acc, Apk);

  // ---- layer 2 projections via WMMA ----
  pack_B<<<(F1 * DD + T - 1) / T, T, 0, stream>>>(W_src2, Ws2p);
  pack_B<<<(F1 * DD + T - 1) / T, T, 0, stream>>>(W_dst2, Wd2p);
  {
    dim3 grid(NN / 16, 3);            // 1250 row tiles x (12 strips / 4 per block)
    dim3 block(32, 4);                // 4 waves; each computes a 16x64 strip
    l2_gemm<<<grid, block, 0, stream>>>(Apk, Ws2p, Wd2p, hs2, hd2);
  }

  // ---- layer 2 edge softmax + aggregation ----
  {
    dim3 block(32, 8);
    unsigned grid = (EE + 8 - 1) / 8;
    l2_score<<<grid, block, 0, stream>>>(edge_src, edge_dst, hs2, hd2, attn2, sc2, m2);
    l2_exp<<<(EE + T - 1) / T, T, 0, stream>>>(edge_dst, sc2, m2, den2);
    l2_agg<<<grid, block, 0, stream>>>(edge_src, edge_dst, hs2, sc2, den2, nodeE);
  }

  // ---- final assembly ----
  {
    long n = (long)BB * SS * DD;
    assemble<<<(unsigned)((n + T - 1) / T), T, 0, stream>>>(
        trj_token, min_list, wd_list, day_list, grid_list,
        nodeE, grid_tbl, dt_tbl, wk_tbl, day_tbl, out);
  }
}
